// ConformerBlockQuant_73924977099289
// MI455X (gfx1250) — 8-bit matrix operand order repaired and checked by reading, compiled, not run on hardware
//
#include <hip/hip_runtime.h>
#include <stdint.h>

typedef __attribute__((ext_vector_type(8))) int v8i;

#define QMINF (-128.0f)
#define QMAXF (127.0f)

__device__ __forceinline__ uint32_t enc_f(float f) {
  uint32_t u = __float_as_uint(f);
  return (u & 0x80000000u) ? ~u : (u | 0x80000000u);
}
__device__ __forceinline__ float dec_f(uint32_t e) {
  uint32_t u = (e & 0x80000000u) ? (e & 0x7fffffffu) : ~e;
  return __uint_as_float(u);
}

__device__ __forceinline__ void qparams(const uint32_t* mn_e, const uint32_t* mx_e,
                                        float& s, float& zp) {
  float mn = fminf(dec_f(*mn_e), 0.0f);
  float mx = fmaxf(dec_f(*mx_e), 0.0f);
  s = fmaxf((mx - mn) * (1.0f / 255.0f), 1e-8f);
  zp = QMINF - rintf(mn / s);
}
__device__ __forceinline__ float fq(float x, float invs, float s, float zp) {
  float q = fminf(fmaxf(rintf(x * invs) + zp, QMINF), QMAXF);
  return (q - zp) * s;
}
__device__ __forceinline__ int quant_i8(float x, float invs, float zp) {
  float q = fminf(fmaxf(rintf(x * invs) + zp, QMINF), QMAXF);
  return (int)q;
}

__device__ __forceinline__ void obs_update(float vmn, float vmx,
                                           uint32_t* gmn, uint32_t* gmx) {
  __shared__ uint32_t smn, smx;
  if (threadIdx.x == 0) { smn = 0xFFFFFFFFu; smx = 0u; }
  __syncthreads();
  atomicMin(&smn, enc_f(vmn));
  atomicMax(&smx, enc_f(vmx));
  __syncthreads();
  if (threadIdx.x == 0) { atomicMin(gmn, smn); atomicMax(gmx, smx); }
}


__global__ void k_init(uint32_t* mn, uint32_t* mx, uint32_t* wmax) {
  int t = threadIdx.x;
  if (t < 8) { mn[t] = 0xFFFFFFFFu; mx[t] = 0u; }
  if (t < 3) wmax[t] = 0u;
}

__global__ void k_minmax(const float* __restrict__ x, long n,
                         uint32_t* gmn, uint32_t* gmx) {
  long i = (long)blockIdx.x * blockDim.x + threadIdx.x;
  long st = (long)gridDim.x * blockDim.x;
  float vmn = INFINITY, vmx = -INFINITY;
  for (; i < n; i += st) { float v = x[i]; vmn = fminf(vmn, v); vmx = fmaxf(vmx, v); }
  obs_update(vmn, vmx, gmn, gmx);
}

__global__ void k_absmax(const float* __restrict__ w, long n, uint32_t* gmx) {
  long i = (long)blockIdx.x * blockDim.x + threadIdx.x;
  long st = (long)gridDim.x * blockDim.x;
  float amx = 0.0f;
  for (; i < n; i += st) amx = fmaxf(amx, fabsf(w[i]));
  __shared__ uint32_t smx;
  if (threadIdx.x == 0) smx = 0u;
  __syncthreads();
  atomicMax(&smx, enc_f(amx));
  __syncthreads();
  if (threadIdx.x == 0) atomicMax(gmx, smx);
}

__global__ void k_quantw(const float* __restrict__ w, int Kw,
                         const uint32_t* wmax_e, int8_t* __restrict__ wq,
                         int* __restrict__ colsum) {
  int n = blockIdx.x;
  float s = fmaxf(dec_f(*wmax_e) * (1.0f / 127.0f), 1e-8f);
  float invs = 1.0f / s;
  __shared__ int ssum;
  if (threadIdx.x == 0) ssum = 0;
  __syncthreads();
  int acc = 0;
  for (int k = threadIdx.x; k < Kw; k += blockDim.x) {
    float q = fminf(fmaxf(rintf(w[(long)n * Kw + k] * invs), QMINF), QMAXF);
    int qi = (int)q;
    wq[(long)n * Kw + k] = (int8_t)qi;
    acc += qi;
  }
  atomicAdd(&ssum, acc);
  __syncthreads();
  if (threadIdx.x == 0) colsum[n] = ssum;
}

__global__ void k_ln(const float* __restrict__ x, const float* __restrict__ g,
                     const float* __restrict__ b,
                     const uint32_t* mn0, const uint32_t* mx0,
                     _Float16* __restrict__ out,
                     uint32_t* gmn, uint32_t* gmx, int F) {
  long row = blockIdx.x;
  float s, zp; qparams(mn0, mx0, s, zp);
  float invs = 1.0f / s;
  const float* xr = x + row * F;
  int base = threadIdx.x * 4;
  float v[4]; float sum = 0.0f;
  for (int j = 0; j < 4; j++) { v[j] = fq(xr[base + j], invs, s, zp); sum += v[j]; }
  __shared__ float red[128];
  red[threadIdx.x] = sum; __syncthreads();
  for (int o = 64; o > 0; o >>= 1) {
    if (threadIdx.x < o) red[threadIdx.x] += red[threadIdx.x + o];
    __syncthreads();
  }
  float mean = red[0] / F;
  __syncthreads();
  float asum = 0.0f;
  for (int j = 0; j < 4; j++) { v[j] -= mean; asum += fabsf(v[j]); }
  red[threadIdx.x] = asum; __syncthreads();
  for (int o = 64; o > 0; o >>= 1) {
    if (threadIdx.x < o) red[threadIdx.x] += red[threadIdx.x + o];
    __syncthreads();
  }
  float inv = 1.0f / (red[0] / F + 1e-5f);
  float vmn = INFINITY, vmx = -INFINITY;
  for (int j = 0; j < 4; j++) {
    float y = v[j] * inv * g[base + j] + b[base + j];
    vmn = fminf(vmn, y); vmx = fmaxf(vmx, y);
    out[row * F + base + j] = (_Float16)y;
  }
  obs_update(vmn, vmx, gmn, gmx);
}

__global__ void k_quant8(const _Float16* __restrict__ in, long n,
                         const uint32_t* mn, const uint32_t* mx,
                         int8_t* __restrict__ out) {
  float s, zp; qparams(mn, mx, s, zp);
  float invs = 1.0f / s;
  long i = (long)blockIdx.x * blockDim.x + threadIdx.x;
  long st = (long)gridDim.x * blockDim.x;
  for (; i < n; i += st) out[i] = (int8_t)quant_i8((float)in[i], invs, zp);
}

__global__ void __launch_bounds__(256)
k_gemm(const int8_t* __restrict__ A, const int8_t* __restrict__ W,
       const int* __restrict__ colsum, const float* __restrict__ bias,
       const uint32_t* amn, const uint32_t* amx, const uint32_t* wmax_e,
       _Float16* __restrict__ Y, uint32_t* gmn, uint32_t* gmx,
       int M, int N, int K) {
  int lane = threadIdx.x & 31;
  int wave = threadIdx.x >> 5;
  int hi = lane >> 4;
  int l = lane & 15;
  int mBase = blockIdx.x * 16;
  int nBase = (blockIdx.y * 8 + wave) * 16;

  const int* ap = (const int*)(A + (long)(mBase + l) * K + 8 * hi);
  const int* bp = (const int*)(W + (long)(nBase + l) * K + 8 * hi);

  v8i acc = {0, 0, 0, 0, 0, 0, 0, 0};
  for (int k0 = 0; k0 < K; k0 += 64) {
    const int* a32 = ap + (k0 >> 2);
    const int* b32 = bp + (k0 >> 2);
    v8i a, b;
    a[0] = a32[0];  a[1] = a32[1];  a[2] = a32[4];  a[3] = a32[5];
    a[4] = a32[8];  a[5] = a32[9];  a[6] = a32[12]; a[7] = a32[13];
    b[0] = b32[0];  b[1] = b32[1];  b[2] = b32[4];  b[3] = b32[5];
    b[4] = b32[8];  b[5] = b32[9];  b[6] = b32[12]; b[7] = b32[13];
    acc = __builtin_amdgcn_wmma_i32_16x16x64_iu8(true, a, true, b, acc, false, false);
  }

  float s, zp; qparams(amn, amx, s, zp);
  float sw = fmaxf(dec_f(*wmax_e) * (1.0f / 127.0f), 1e-8f);
  float cs = s * sw;
  int zpi = (int)zp;
  int n = nBase + l;
  int corr = zpi * colsum[n];
  float bn = bias[n];
  float vmn = INFINITY, vmx = -INFINITY;
  for (int v = 0; v < 8; v++) {
    int m = mBase + v + 8 * hi;
    float val = cs * (float)(acc[v] - corr) + bn;
    vmn = fminf(vmn, val); vmx = fmaxf(vmx, val);
    Y[(long)m * N + n] = (_Float16)val;
  }
  obs_update(vmn, vmx, gmn, gmx);
}

__global__ void k_glu(const _Float16* __restrict__ y, long rows, int F,
                      const uint32_t* mn2, const uint32_t* mx2,
                      _Float16* __restrict__ out, uint32_t* gmn, uint32_t* gmx) {
  float s, zp; qparams(mn2, mx2, s, zp);
  float invs = 1.0f / s;
  long n = rows * F;
  int W2 = 2 * F;
  float vmn = INFINITY, vmx = -INFINITY;
  long i = (long)blockIdx.x * blockDim.x + threadIdx.x;
  long st = (long)gridDim.x * blockDim.x;
  for (; i < n; i += st) {
    long r = i / F; int f = (int)(i - r * F);
    float av = fq((float)y[r * W2 + f], invs, s, zp);
    float gv = fq((float)y[r * W2 + F + f], invs, s, zp);
    float val = av * (1.0f / (1.0f + __expf(-gv)));
    vmn = fminf(vmn, val); vmx = fmaxf(vmx, val);
    out[i] = (_Float16)val;
  }
  obs_update(vmn, vmx, gmn, gmx);
}

__global__ void k_qtrans(const _Float16* __restrict__ in,
                         const uint32_t* mn, const uint32_t* mx,
                         int8_t* __restrict__ out, int B, int T, int F) {
  float s, zp; qparams(mn, mx, s, zp);
  float invs = 1.0f / s;
  long n = (long)B * T * F;
  long i = (long)blockIdx.x * blockDim.x + threadIdx.x;
  long st = (long)gridDim.x * blockDim.x;
  for (; i < n; i += st) {
    long b = i / ((long)F * T);
    long rem = i - b * (long)F * T;
    int f = (int)(rem / T);
    int t = (int)(rem - (long)f * T);
    out[i] = (int8_t)quant_i8((float)in[(b * T + t) * (long)F + f], invs, zp);
  }
}

__global__ void k_dwconv(const int8_t* __restrict__ a, const int8_t* __restrict__ wq,
                         const int* __restrict__ wsum, const float* __restrict__ bias,
                         const uint32_t* mn3, const uint32_t* mx3, const uint32_t* wmax_e,
                         _Float16* __restrict__ out, uint32_t* gmn, uint32_t* gmx,
                         int B, int F, int T, int Kw) {
  float s, zp; qparams(mn3, mx3, s, zp);
  int zpi = (int)zp;
  float sw = fmaxf(dec_f(*wmax_e) * (1.0f / 127.0f), 1e-8f);
  float cs = s * sw;
  int pad = (Kw - 1) / 2;
  long n = (long)B * F * T;
  float vmn = INFINITY, vmx = -INFINITY;
  long i = (long)blockIdx.x * blockDim.x + threadIdx.x;
  long st = (long)gridDim.x * blockDim.x;
  for (; i < n; i += st) {
    long bf = i / T; int t = (int)(i - bf * T);
    int c = (int)(bf % F);
    const int8_t* row = a + bf * T;
    int acc = 0;
    for (int k = 0; k < Kw; k++) {
      int tt = t + k - pad;
      int q = (tt >= 0 && tt < T) ? (int)row[tt] : zpi;
      acc += q * (int)wq[c * Kw + k];
    }
    float val = cs * (float)(acc - zpi * wsum[c]) + bias[c];
    vmn = fminf(vmn, val); vmx = fmaxf(vmx, val);
    out[i] = (_Float16)val;
  }
  obs_update(vmn, vmx, gmn, gmx);
}

__global__ void k_bn(const _Float16* __restrict__ in,
                     const uint32_t* mn4, const uint32_t* mx4,
                     const float* __restrict__ gamma, const float* __restrict__ beta,
                     const float* __restrict__ mean, const float* __restrict__ var,
                     _Float16* __restrict__ out, uint32_t* gmn, uint32_t* gmx,
                     int F, int T, long n) {
  float s, zp; qparams(mn4, mx4, s, zp);
  float invs = 1.0f / s;
  float vmn = INFINITY, vmx = -INFINITY;
  long i = (long)blockIdx.x * blockDim.x + threadIdx.x;
  long st = (long)gridDim.x * blockDim.x;
  for (; i < n; i += st) {
    int c = (int)((i / T) % F);
    float v = fq((float)in[i], invs, s, zp);
    float val = (v - mean[c]) * (gamma[c] * rsqrtf(var[c] + 1e-5f)) + beta[c];
    vmn = fminf(vmn, val); vmx = fmaxf(vmx, val);
    out[i] = (_Float16)val;
  }
  obs_update(vmn, vmx, gmn, gmx);
}

__global__ void k_silu(const _Float16* __restrict__ in, long n,
                       const uint32_t* mn5, const uint32_t* mx5,
                       _Float16* __restrict__ out, uint32_t* gmn, uint32_t* gmx) {
  float s, zp; qparams(mn5, mx5, s, zp);
  float invs = 1.0f / s;
  float vmn = INFINITY, vmx = -INFINITY;
  long i = (long)blockIdx.x * blockDim.x + threadIdx.x;
  long st = (long)gridDim.x * blockDim.x;
  for (; i < n; i += st) {
    float v = fq((float)in[i], invs, s, zp);
    float val = v * (1.0f / (1.0f + __expf(-v)));
    vmn = fminf(vmn, val); vmx = fmaxf(vmx, val);
    out[i] = (_Float16)val;
  }
  obs_update(vmn, vmx, gmn, gmx);
}

__global__ void k_qtransback(const _Float16* __restrict__ in,
                             const uint32_t* mn, const uint32_t* mx,
                             int8_t* __restrict__ out, int B, int T, int F) {
  float s, zp; qparams(mn, mx, s, zp);
  float invs = 1.0f / s;
  long n = (long)B * T * F;
  long i = (long)blockIdx.x * blockDim.x + threadIdx.x;
  long st = (long)gridDim.x * blockDim.x;
  for (; i < n; i += st) {
    long b = i / ((long)T * F);
    long rem = i - b * (long)T * F;
    int t = (int)(rem / F);
    int f = (int)(rem - (long)t * F);
    out[i] = (int8_t)quant_i8((float)in[(b * F + f) * (long)T + t], invs, zp);
  }
}

__global__ void k_final(const _Float16* __restrict__ in, long n,
                        const uint32_t* mn7, const uint32_t* mx7,
                        float* __restrict__ out) {
  float s, zp; qparams(mn7, mx7, s, zp);
  float invs = 1.0f / s;
  long i = (long)blockIdx.x * blockDim.x + threadIdx.x;
  long st = (long)gridDim.x * blockDim.x;
  for (; i < n; i += st) out[i] = fq((float)in[i], invs, s, zp);
}


static inline int gs(long n) {
  long b = (n + 255) / 256;
  return (int)(b < 8192 ? b : 8192);
}

extern "C" void kernel_launch(void* const* d_in, const int* in_sizes, int n_in,
                              void* d_out, int out_size, void* d_ws, size_t ws_size,
                              hipStream_t stream) {
  const float* x        = (const float*)d_in[0];
  const float* ln_scale = (const float*)d_in[1];
  const float* ln_bias  = (const float*)d_in[2];
  const float* w1       = (const float*)d_in[3];
  const float* b1       = (const float*)d_in[4];
  const float* dw_w     = (const float*)d_in[5];
  const float* dw_b     = (const float*)d_in[6];
  const float* bn_gamma = (const float*)d_in[7];
  const float* bn_beta  = (const float*)d_in[8];
  const float* bn_mean  = (const float*)d_in[9];
  const float* bn_var   = (const float*)d_in[10];
  const float* w2       = (const float*)d_in[11];
  const float* b2       = (const float*)d_in[12];

  const int B = 32, T = 1024, F = 512, KW = 31;
  const long M = (long)B * T;
  const long NEL = M * F;

  uint8_t* wsb = (uint8_t*)d_ws;
  uint32_t* mn   = (uint32_t*)wsb;
  uint32_t* mx   = mn + 8;
  uint32_t* wmax = mx + 8;
  int* colsum1 = (int*)(wsb + 256);
  int* colsum2 = (int*)(wsb + 4608);
  int* dwsum   = (int*)(wsb + 6656);
  int8_t* w1q  = (int8_t*)(wsb + 16384);
  int8_t* w2q  = (int8_t*)(wsb + 16384 + 524288);
  int8_t* dwq  = (int8_t*)(wsb + 16384 + 524288 + 262144);
  _Float16* BUFA = (_Float16*)(wsb + (1ul << 20));
  _Float16* BUFB = (_Float16*)(wsb + (1ul << 20) + (64ul << 20));
  int8_t*   BUFC = (int8_t*)(wsb + (1ul << 20) + (96ul << 20));

  k_init<<<1, 32, 0, stream>>>(mn, mx, wmax);

  k_minmax<<<gs(NEL), 256, 0, stream>>>(x, NEL, mn + 0, mx + 0);
  k_absmax<<<gs((long)2 * F * F), 256, 0, stream>>>(w1, (long)2 * F * F, wmax + 0);
  k_absmax<<<gs((long)F * F), 256, 0, stream>>>(w2, (long)F * F, wmax + 1);
  k_absmax<<<gs((long)F * KW), 256, 0, stream>>>(dw_w, (long)F * KW, wmax + 2);

  k_quantw<<<2 * F, 128, 0, stream>>>(w1, F, wmax + 0, w1q, colsum1);
  k_quantw<<<F, 128, 0, stream>>>(w2, F, wmax + 1, w2q, colsum2);
  k_quantw<<<F, 64, 0, stream>>>(dw_w, KW, wmax + 2, dwq, dwsum);

  k_ln<<<(int)M, 128, 0, stream>>>(x, ln_scale, ln_bias, mn + 0, mx + 0,
                                   BUFB, mn + 1, mx + 1, F);
  k_quant8<<<gs(NEL), 256, 0, stream>>>(BUFB, NEL, mn + 1, mx + 1, BUFC);

  k_gemm<<<dim3((int)(M / 16), (2 * F) / 128), 256, 0, stream>>>(
      BUFC, w1q, colsum1, b1, mn + 1, mx + 1, wmax + 0,
      BUFA, mn + 2, mx + 2, (int)M, 2 * F, F);

  k_glu<<<gs(NEL), 256, 0, stream>>>(BUFA, M, F, mn + 2, mx + 2,
                                     BUFB, mn + 3, mx + 3);
  k_qtrans<<<gs(NEL), 256, 0, stream>>>(BUFB, mn + 3, mx + 3, BUFC, B, T, F);

  k_dwconv<<<gs(NEL), 256, 0, stream>>>(BUFC, dwq, dwsum, dw_b,
                                        mn + 3, mx + 3, wmax + 2,
                                        BUFA, mn + 4, mx + 4, B, F, T, KW);
  k_bn<<<gs(NEL), 256, 0, stream>>>(BUFA, mn + 4, mx + 4, bn_gamma, bn_beta,
                                    bn_mean, bn_var, BUFB, mn + 5, mx + 5,
                                    F, T, NEL);
  k_silu<<<gs(NEL), 256, 0, stream>>>(BUFB, NEL, mn + 5, mx + 5,
                                      BUFA, mn + 6, mx + 6);
  k_qtransback<<<gs(NEL), 256, 0, stream>>>(BUFA, mn + 6, mx + 6, BUFC, B, T, F);

  k_gemm<<<dim3((int)(M / 16), F / 128), 256, 0, stream>>>(
      BUFC, w2q, colsum2, b2, mn + 6, mx + 6, wmax + 1,
      BUFB, mn + 7, mx + 7, (int)M, F, F);

  k_final<<<gs(NEL), 256, 0, stream>>>(BUFB, NEL, mn + 7, mx + 7, (float*)d_out);
}
